// Multi_Heads_Self_Attn_1D_1812476199802
// MI455X (gfx1250) — matrix operand order repaired and checked by reading, not compiled, not run on hardware
//
#include <hip/hip_runtime.h>

// ---------------------------------------------------------------------------
// Multi-head self-attention 1D (linear attention) for MI455X / gfx1250.
//   (QK^T/dk)V  ==  Q(K^T V)/dk   -> per-head 64x64 inner matrix, no LxL tensor
// All GEMM stages via v_wmma_f32_16x16x32_bf16 (f32 accumulate).
// LDS is staged so every fragment read is a ds_load_b128 (no scalar gathers).
// ---------------------------------------------------------------------------

typedef __bf16 bf16_t;
typedef __attribute__((ext_vector_type(16))) __bf16 v16bf;
typedef __attribute__((ext_vector_type(8)))  float  v8f;

typedef union { uint4 q[2]; v16bf v; } frag_u;   // 32B fragment = 2x b128

#define WMMA_BF16(a, b, c) \
  __builtin_amdgcn_wmma_f32_16x16x32_bf16(false, (a), false, (b), (short)0, (c), false, false)

constexpr int Bn  = 16;
constexpr int Cc  = 256;
constexpr int Ll  = 1024;
constexpr int Hh  = 8;
constexpr int DKk = 64;
constexpr int HDd = 512;   // Hh * DKk

// -------------------------------- fp32 -> bf16 -----------------------------
__global__ void __launch_bounds__(256)
cvt_f32_bf16(const float* __restrict__ in, bf16_t* __restrict__ out, int n)
{
  int i = (blockIdx.x * 256 + threadIdx.x) * 4;
  if (i + 3 < n) {
    float4 v = *(const float4*)(in + i);
    out[i + 0] = (bf16_t)v.x;
    out[i + 1] = (bf16_t)v.y;
    out[i + 2] = (bf16_t)v.z;
    out[i + 3] = (bf16_t)v.w;
  }
}

// ------------------------ generic bf16 WMMA GEMM ---------------------------
// D[M,N] = A[M,K] * B[K,N]; A shared across batch, B/D strided per batch.
// Block tile 128x128, 8 waves, each wave a 16x128 strip (8 accumulators).
// A tile row-major (pad 40 -> conflict-free b128 reads); B tile staged in
// WMMA fragment order so each wave reads fragments as 2x ds_load_b128.
// Register double-buffering hides global latency behind the WMMA stream.
template <typename OutT>
__global__ void __launch_bounds__(256)
gemm_bf16_wmma(const bf16_t* __restrict__ A,
               const bf16_t* __restrict__ Bm,
               OutT* __restrict__ D,
               int M, int N, int K,
               long strideB, long strideD)
{
  __shared__ bf16_t As[128][40];        // 128x32 used (80B pitch, 16B aligned)
  __shared__ bf16_t Bf[8 * 32 * 16];    // fragment order: ((nt*32+lane)*16+i)

  const int tid  = threadIdx.x;
  const int wave = tid >> 5;
  const int lane = tid & 31;
  const int row0 = blockIdx.y * 128;
  const int col0 = blockIdx.x * 128;
  const bf16_t* Bb = Bm + strideB * (long)blockIdx.z;
  OutT*         Db = D  + strideD * (long)blockIdx.z;

  const int la = lane & 15;
  const int kb = (lane >> 4) << 4;      // 0 or 16 (A-frag K base)

  v8f acc[8] = {};
  uint4 ga[2], gb[2];

  auto loadG = [&](int k0) {
#pragma unroll
    for (int rep = 0; rep < 2; rep++) {
      int e  = tid * 8 + rep * 2048;
      int ra = e >> 5, ca = e & 31;
      ga[rep] = *(const uint4*)&A[(long)(row0 + ra) * K + k0 + ca];
      int rb = e >> 7, cb = e & 127;
      gb[rep] = *(const uint4*)&Bb[(long)(k0 + rb) * N + col0 + cb];
    }
  };
  auto storeL = [&]() {
#pragma unroll
    for (int rep = 0; rep < 2; rep++) {
      int e  = tid * 8 + rep * 2048;
      int ra = e >> 5, ca = e & 31;
      *(uint4*)&As[ra][ca] = ga[rep];
      int rb = e >> 7, cb = e & 127;
      const bf16_t* src = (const bf16_t*)&gb[rep];
      const int i     = rb & 15;
      const int lhalf = ((rb >> 4) & 1) << 4;
#pragma unroll
      for (int j = 0; j < 8; j++) {         // scatter into fragment order
        int c  = cb + j;
        int nt = c >> 4;
        int ln = (c & 15) + lhalf;
        Bf[((nt << 5) + ln) * 16 + i] = src[j];
      }
    }
  };

  loadG(0);
  for (int k0 = 0; k0 < K; k0 += 32) {
    storeL();
    __syncthreads();
    if (k0 + 32 < K) loadG(k0 + 32);    // prefetch next tile into VGPRs

    frag_u a;                           // contiguous: 2x ds_load_b128
    const int ar = (wave << 4) + la;
    a.q[0] = *(const uint4*)&As[ar][kb];
    a.q[1] = *(const uint4*)&As[ar][kb + 8];
#pragma unroll
    for (int nt = 0; nt < 8; nt++) {
      frag_u b;
      const bf16_t* bp = &Bf[((nt << 5) + lane) * 16];
      b.q[0] = *(const uint4*)bp;
      b.q[1] = *(const uint4*)(bp + 8);
      acc[nt] = WMMA_BF16(a.v, b.v, acc[nt]);
    }
    __syncthreads();
  }

  // C/D layout: VGPR v -> row v (lanes 0-15) or v+8 (lanes 16-31), col = lane&15
  const int rbase = row0 + (wave << 4) + ((lane >> 4) << 3);
#pragma unroll
  for (int nt = 0; nt < 8; nt++) {
    const int cc = col0 + (nt << 4) + la;
#pragma unroll
    for (int v = 0; v < 8; v++)
      Db[(long)(rbase + v) * N + cc] = (OutT)acc[nt][v];
  }
}

// ---------------- Mt = (V_h @ K_h^T) / DK == (K^T V)^T  per (b,h) ----------
// Computing the transposed inner matrix directly makes the next stage's
// A operand row-major (contiguous LDS fragment reads).
__global__ void __launch_bounds__(128)
kv_wmma(const bf16_t* __restrict__ QKV, bf16_t* __restrict__ Mt)
{
  const int bh = blockIdx.x;                       // b*H + h
  const int b = bh >> 3, h = bh & 7;
  const bf16_t* Kh = QKV + ((long)b * 3 * HDd + HDd     + h * DKk) * Ll;
  const bf16_t* Vh = QKV + ((long)b * 3 * HDd + 2 * HDd + h * DKk) * Ll;

  __shared__ bf16_t Vs[64][40];
  __shared__ bf16_t Ks[64][40];

  const int tid = threadIdx.x, wave = tid >> 5, lane = tid & 31;
  const int la = lane & 15;
  const int kb  = (lane >> 4) << 3;
  const int kb2 = (lane >> 4) << 4;

  v8f acc[4] = {};
  uint4 gk[2], gv[2];

  auto loadG = [&](int l0) {
#pragma unroll
    for (int rep = 0; rep < 2; rep++) {
      int e = tid * 8 + rep * 1024;
      int r = e >> 5, c = e & 31;
      gk[rep] = *(const uint4*)&Kh[(long)r * Ll + l0 + c];
      gv[rep] = *(const uint4*)&Vh[(long)r * Ll + l0 + c];
    }
  };

  loadG(0);
  for (int l0 = 0; l0 < Ll; l0 += 32) {
#pragma unroll
    for (int rep = 0; rep < 2; rep++) {
      int e = tid * 8 + rep * 1024;
      int r = e >> 5, c = e & 31;
      *(uint4*)&Ks[r][c] = gk[rep];
      *(uint4*)&Vs[r][c] = gv[rep];
    }
    __syncthreads();
    if (l0 + 32 < Ll) loadG(l0 + 32);

    frag_u a;                                   // A = V_h rows (contiguous)
    const int ar = (wave << 4) + la;
    a.q[0] = *(const uint4*)&Vs[ar][kb];
    a.q[1] = *(const uint4*)&Vs[ar][16 + kb];
#pragma unroll
    for (int nt = 0; nt < 4; nt++) {
      frag_u bfr;                               // B[k][n] = K_h[n][l0+k] (contiguous)
      const int bc = (nt << 4) + la;
      bfr.q[0] = *(const uint4*)&Ks[bc][kb];
      bfr.q[1] = *(const uint4*)&Ks[bc][16 + kb];
      acc[nt] = WMMA_BF16(a.v, bfr.v, acc[nt]);
    }
    __syncthreads();
  }

  bf16_t* Mp = Mt + (long)bh * 64 * 64;
  const int rbase = (wave << 4) + ((lane >> 4) << 3);
#pragma unroll
  for (int nt = 0; nt < 4; nt++)
#pragma unroll
    for (int v = 0; v < 8; v++)                 // fold 1/DK here
      Mp[(rbase + v) * 64 + (nt << 4) + la] = (bf16_t)(acc[nt][v] * (1.0f / 64.0f));
}

// ------------------- Out_h = Mt @ Q_h  (64 x 128-col tile) -----------------
__global__ void __launch_bounds__(256)
qm_wmma(const bf16_t* __restrict__ QKV, const bf16_t* __restrict__ Mt,
        bf16_t* __restrict__ Out)
{
  const int bh   = blockIdx.y;
  const int b = bh >> 3, h = bh & 7;
  const int col0 = blockIdx.x * 128;
  const bf16_t* Qh = QKV + ((long)b * 3 * HDd + h * DKk) * Ll;
  const bf16_t* Mp = Mt  + (long)bh * 64 * 64;
  bf16_t*       Oh = Out + (long)bh * DKk * Ll;

  __shared__ bf16_t Ms[64][72];           // row-major Mt (144B pitch, aligned)
  __shared__ bf16_t Qf[2 * 8 * 32 * 16];  // fragment order, both K-steps

  const int tid = threadIdx.x, wave = tid >> 5, lane = tid & 31;

  // stage Mt rows (vector copies)
#pragma unroll
  for (int rep = 0; rep < 2; rep++) {
    int e = tid * 8 + rep * 2048;
    *(uint4*)&Ms[e >> 6][e & 63] = *(const uint4*)&Mp[e];
  }
  // stage Q tile into fragment order: ((step*8+nt)*32+lane)*16+i
#pragma unroll
  for (int rep = 0; rep < 4; rep++) {
    int e  = tid * 8 + rep * 2048;
    int kk = e >> 7, cc = e & 127;
    uint4 gq = *(const uint4*)&Qh[(long)kk * Ll + col0 + cc];
    const bf16_t* src = (const bf16_t*)&gq;
    const int step  = kk >> 5;
    const int i     = kk & 15;
    const int lhalf = ((kk >> 4) & 1) << 4;
#pragma unroll
    for (int j = 0; j < 8; j++) {
      int c  = cc + j;
      int nt = c >> 4;
      int ln = (c & 15) + lhalf;
      Qf[(((step << 3) + nt) << 5 | ln) * 16 + i] = src[j];
    }
  }
  __syncthreads();

  const int strip = wave & 3;                // output row strip (d)
  const int ntoff = (wave >> 2) << 2;        // column-tile offset: 0 or 4
  const int la = lane & 15;
  const int kb = (lane >> 4) << 4;

  v8f acc[4] = {};
#pragma unroll
  for (int s = 0; s < 2; s++) {              // K-steps (K = 64)
    frag_u a;                                // A = Mt rows (contiguous)
    const int ar = (strip << 4) + la;
    const int k0 = s << 5;
    a.q[0] = *(const uint4*)&Ms[ar][k0 + kb];
    a.q[1] = *(const uint4*)&Ms[ar][k0 + kb + 8];
#pragma unroll
    for (int nt = 0; nt < 4; nt++) {
      frag_u bfr;
      const bf16_t* bp = &Qf[(((s << 3) + ntoff + nt) << 5 | lane) * 16];
      bfr.q[0] = *(const uint4*)bp;
      bfr.q[1] = *(const uint4*)(bp + 8);
      acc[nt] = WMMA_BF16(a.v, bfr.v, acc[nt]);
    }
  }

  const int rbase = (strip << 4) + ((lane >> 4) << 3);
#pragma unroll
  for (int nt = 0; nt < 4; nt++) {
    const int cc = col0 + ((ntoff + nt) << 4) + la;
#pragma unroll
    for (int v = 0; v < 8; v++)
      Oh[(long)(rbase + v) * Ll + cc] = (bf16_t)acc[nt][v];
  }
}

// -------------- InstanceNorm(L) + gamma residual + LeakyReLU ---------------
__global__ void __launch_bounds__(256)
inorm_res_leaky(const float* __restrict__ O, const float* __restrict__ x,
                const float* __restrict__ gamma, float* __restrict__ y)
{
  const long row = blockIdx.x;                 // b*C + c
  const float* orow = O + row * Ll;
  const float* xrow = x + row * Ll;
  float*       yrow = y + row * Ll;
  const int tid = threadIdx.x;

  float4 v = *(const float4*)(orow + tid * 4);
  float s  = v.x + v.y + v.z + v.w;
  float s2 = v.x * v.x + v.y * v.y + v.z * v.z + v.w * v.w;

  __shared__ float rs[256], rq[256];
  rs[tid] = s; rq[tid] = s2;
  __syncthreads();
  for (int off = 128; off > 0; off >>= 1) {
    if (tid < off) { rs[tid] += rs[tid + off]; rq[tid] += rq[tid + off]; }
    __syncthreads();
  }
  const float mu  = rs[0] * (1.0f / Ll);
  const float var = rq[0] * (1.0f / Ll) - mu * mu;     // population var (ddof=0)
  const float inv = rsqrtf(var + 1e-5f);
  const float g   = gamma[0];

  float4 xv = *(const float4*)(xrow + tid * 4);
  float4 o;
  o.x = (v.x - mu) * inv * g + xv.x;
  o.y = (v.y - mu) * inv * g + xv.y;
  o.z = (v.z - mu) * inv * g + xv.z;
  o.w = (v.w - mu) * inv * g + xv.w;
  o.x = o.x >= 0.f ? o.x : 0.01f * o.x;
  o.y = o.y >= 0.f ? o.y : 0.01f * o.y;
  o.z = o.z >= 0.f ? o.z : 0.01f * o.z;
  o.w = o.w >= 0.f ? o.w : 0.01f * o.w;
  *(float4*)(yrow + tid * 4) = o;
}

// ---------------------------------------------------------------------------
extern "C" void kernel_launch(void* const* d_in, const int* in_sizes, int n_in,
                              void* d_out, int out_size, void* d_ws, size_t ws_size,
                              hipStream_t stream)
{
  const float* x     = (const float*)d_in[0];   // [B,C,L]
  const float* Wq    = (const float*)d_in[1];   // [HD,C]
  const float* Wk    = (const float*)d_in[2];
  const float* Wv    = (const float*)d_in[3];
  const float* Wo    = (const float*)d_in[4];   // [C,HD]
  const float* gamma = (const float*)d_in[5];
  float* y = (float*)d_out;

  // workspace carve-up (256B aligned); total ~94 MB
  char* wsp = (char*)d_ws;
  auto alloc = [&](size_t bytes) -> char* {
    char* p = wsp;
    wsp += (bytes + 255) & ~(size_t)255;
    return p;
  };
  bf16_t* xb      = (bf16_t*)alloc((size_t)Bn * Cc * Ll * 2);
  bf16_t* wqkvb   = (bf16_t*)alloc((size_t)3 * HDd * Cc * 2);     // [Wq;Wk;Wv]
  bf16_t* wob     = (bf16_t*)alloc((size_t)Cc * HDd * 2);
  bf16_t* qkvb    = (bf16_t*)alloc((size_t)Bn * 3 * HDd * Ll * 2); // [B][3HD][L]
  bf16_t* Mtb     = (bf16_t*)alloc((size_t)Bn * Hh * 64 * 64 * 2);
  bf16_t* mergedb = (bf16_t*)alloc((size_t)Bn * HDd * Ll * 2);
  float*  Of      = (float*)alloc((size_t)Bn * Cc * Ll * 4);

  // 1) fp32 -> bf16 conversions (Wq/Wk/Wv concatenated into one operand)
  auto cvt = [&](const float* in, bf16_t* out, int n) {
    cvt_f32_bf16<<<dim3((n / 4 + 255) / 256), dim3(256), 0, stream>>>(in, out, n);
  };
  cvt(x,  xb, Bn * Cc * Ll);
  cvt(Wq, wqkvb + 0 * (size_t)HDd * Cc, HDd * Cc);
  cvt(Wk, wqkvb + 1 * (size_t)HDd * Cc, HDd * Cc);
  cvt(Wv, wqkvb + 2 * (size_t)HDd * Cc, HDd * Cc);
  cvt(Wo, wob, Cc * HDd);

  // 2) fused QKV = [Wq;Wk;Wv] @ x_b : [1536,256] x [256,L] per batch
  {
    dim3 grid(Ll / 128, (3 * HDd) / 128, Bn), blk(256);
    gemm_bf16_wmma<bf16_t><<<grid, blk, 0, stream>>>(
        wqkvb, xb, qkvb, 3 * HDd, Ll, Cc, (long)Cc * Ll, (long)3 * HDd * Ll);
  }

  // 3) Mt = (V_h K_h^T)/DK per (b,h): 64x64 with K-dim = L
  kv_wmma<<<dim3(Bn * Hh), dim3(128), 0, stream>>>(qkvb, Mtb);

  // 4) merged_h = Mt @ Q_h : [64,L] per (b,h)
  qm_wmma<<<dim3(Ll / 128, Bn * Hh), dim3(256), 0, stream>>>(qkvb, Mtb, mergedb);

  // 5) O = Wo @ merged_b : [C,HD] x [HD,L] -> [C,L] fp32 per batch
  {
    dim3 grid(Ll / 128, Cc / 128, Bn), blk(256);
    gemm_bf16_wmma<float><<<grid, blk, 0, stream>>>(
        wob, mergedb, Of, Cc, Ll, HDd, (long)HDd * Ll, (long)Cc * Ll);
  }

  // 6) InstanceNorm over L + gamma residual + LeakyReLU
  inorm_res_leaky<<<dim3(Bn * Cc), dim3(256), 0, stream>>>(Of, x, gamma, y);
}
